// Domain_adaptation_41274635715290
// MI455X (gfx1250) — matrix operand order repaired and checked by reading, not compiled, not run on hardware
//
#include <hip/hip_runtime.h>
#include <math.h>

// ---------------- CDNA5 WMMA types ----------------
typedef __attribute__((ext_vector_type(16))) __bf16 v16bf;
typedef __attribute__((ext_vector_type(8)))  float  v8f;

#define NROW 1280
#define FDIM 32768
#define NREG 10
#define NLOC 128   // rows per region

// ---------------- helpers ----------------
__device__ inline unsigned short f2bf(float f) {
  unsigned int u = __float_as_uint(f);
  u += 0x7FFFu + ((u >> 16) & 1u);      // round-to-nearest-even
  return (unsigned short)(u >> 16);
}

// A fragment, 16x32 bf16 (MxK), A row-major [rows x K].
// Lane L (<16): row=L, frag[0..7]=K[0..7], frag[8..15]=K[16..23]
// Lane L (>=16): row=L-16, frag[0..7]=K[8..15], frag[8..15]=K[24..31]
__device__ inline v16bf ld_a(const unsigned short* p) {
  union { v16bf v; uint4 u[2]; } r;
  r.u[0] = *(const uint4*)(p);
  r.u[1] = *(const uint4*)(p + 16);
  return r.v;
}
// B fragment, 32x16 bf16 (KxN) for A*B^T with B row-major [cols x K].
// Lane L: col=L&15, frag[0..7]=K[(L>>4)*8..+7], frag[8..15]=K[16+(L>>4)*8..+7].
__device__ inline v16bf ld_b(const unsigned short* p) {
  union { v16bf v; uint4 u[2]; } r;
  r.u[0] = *(const uint4*)(p);
  r.u[1] = *(const uint4*)(p + 16);
  return r.v;
}

__device__ inline float wave_sum(float v) {
  for (int m = 16; m >= 1; m >>= 1) v += __shfl_xor(v, m, 32);
  return v;
}

// ---------------- kernel 0: zero accumulators ----------------
__global__ void zero_accum_kernel(float* accum) {
  if (threadIdx.x < 8) accum[threadIdx.x] = 0.f;
}

// ---------------- kernel 1: f32 -> bf16 features + row norms ----------------
__global__ __launch_bounds__(256) void feat_prep_kernel(
    const float* __restrict__ src, const float* __restrict__ tgt,
    unsigned short* __restrict__ sB, unsigned short* __restrict__ tB,
    float* __restrict__ s2, float* __restrict__ t2) {
  int row = blockIdx.x;
  const float* in; unsigned short* out; float* nrm; int r;
  if (row < NROW) { r = row;        in = src + (size_t)r * FDIM; out = sB + (size_t)r * FDIM; nrm = s2; }
  else            { r = row - NROW; in = tgt + (size_t)r * FDIM; out = tB + (size_t)r * FDIM; nrm = t2; }
  const float4* in4 = (const float4*)in;
  ushort4* out4 = (ushort4*)out;
  float ss = 0.f;
  for (int i = threadIdx.x; i < FDIM / 4; i += 256) {
    float4 v = in4[i];
    ss += v.x * v.x + v.y * v.y + v.z * v.z + v.w * v.w;
    out4[i] = make_ushort4(f2bf(v.x), f2bf(v.y), f2bf(v.z), f2bf(v.w));
  }
  ss = wave_sum(ss);
  __shared__ float red[8];
  if ((threadIdx.x & 31) == 0) red[threadIdx.x >> 5] = ss;
  __syncthreads();
  if (threadIdx.x == 0) {
    float t = 0.f;
    for (int w = 0; w < 8; ++w) t += red[w];
    nrm[r] = t;
  }
}

// ---------------- kernel 2: transpose+convert W [F x N] f32 -> Wt [N x F] bf16 ----------------
__global__ __launch_bounds__(256) void transpose_convert_kernel(
    const float* __restrict__ W, unsigned short* __restrict__ Wt, int F, int N) {
  __shared__ float tile[32][33];
  int kb = blockIdx.x * 32, jb = blockIdx.y * 32;
  int tx = threadIdx.x, ty = threadIdx.y;          // block (32,8)
  for (int r = ty; r < 32; r += 8)
    tile[r][tx] = W[(size_t)(kb + r) * N + (jb + tx)];
  __syncthreads();
  for (int jl = ty; jl < 32; jl += 8)
    Wt[(size_t)(jb + jl) * F + (kb + tx)] = f2bf(tile[tx][jl]);
}

// ---------------- kernel 3: WMMA gram + exp + reduce ----------------
// 4 waves per block, each wave owns a 64x64 tile (4x4 WMMA frags) of a
// 128x128 macro tile; 16 wmma per 16 b128 loads per K-step.
__device__ inline float tile_exp_sum(v8f c, int gi0, int gj,
                                     const float* a2, const float* b2, int sym) {
  float bj = b2[gj];
  float s = 0.f;
#pragma unroll
  for (int r = 0; r < 8; ++r) {
    int gi = gi0 + r;
    float e = (sym && gi == gj) ? 1.0f : expf(c[r] - 0.5f * (a2[gi] + bj));
    s += e;
  }
  return s;
}

__global__ __launch_bounds__(128) void gram_exp_kernel(
    const unsigned short* __restrict__ A, const unsigned short* __restrict__ B,
    const float* __restrict__ a2, const float* __restrict__ b2,
    int K, int regionStride, int sym, float* slot) {
  const int lane = threadIdx.x & 31;
  const int wid  = threadIdx.x >> 5;          // 4 waves: 2x2 of 64x64 tiles
  const int regOff = blockIdx.z * regionStride;
  const int rBase = regOff + blockIdx.x * 128 + (wid >> 1) * 64;
  const int cBase = regOff + blockIdx.y * 128 + (wid & 1) * 64;

  const unsigned short* pa = A + (size_t)(rBase + (lane & 15)) * K + ((lane >> 4) << 3);
  const unsigned short* pb = B + (size_t)(cBase + (lane & 15)) * K + ((lane >> 4) << 3);
  const size_t rowStep16 = (size_t)16 * K;

  v8f acc[4][4];
#pragma unroll
  for (int i = 0; i < 4; ++i)
#pragma unroll
    for (int j = 0; j < 4; ++j) acc[i][j] = (v8f){};

  for (int k = 0; k < K; k += 32) {
    __builtin_prefetch(pa + k + 256, 0, 1);
    __builtin_prefetch(pb + k + 256, 0, 1);
    v16bf a[4], b[4];
#pragma unroll
    for (int i = 0; i < 4; ++i) a[i] = ld_a(pa + i * rowStep16 + k);
#pragma unroll
    for (int j = 0; j < 4; ++j) b[j] = ld_b(pb + j * rowStep16 + k);
#pragma unroll
    for (int i = 0; i < 4; ++i)
#pragma unroll
      for (int j = 0; j < 4; ++j)
        acc[i][j] = __builtin_amdgcn_wmma_f32_16x16x32_bf16(
            false, a[i], false, b[j], (short)0, acc[i][j], false, false);
  }

  const int iOff = (lane >> 4) << 3;   // C layout: VGPR r -> M=r (lanes 0-15) / r+8 (16-31)
  const int jOff = lane & 15;
  float local = 0.f;
#pragma unroll
  for (int i = 0; i < 4; ++i)
#pragma unroll
    for (int j = 0; j < 4; ++j)
      local += tile_exp_sum(acc[i][j], rBase + i * 16 + iOff,
                            cBase + j * 16 + jOff, a2, b2, sym);
  local = wave_sum(local);
  if (lane == 0) atomicAdd(slot, local);
}

// ---------------- kernel 4: WMMA GEMM with bias (+optional relu) ----------------
__device__ inline void store_tile(v8f c, int i0, int j, const float* bias,
                                  float* out, int ldo, int relu) {
  float b = bias[j];
#pragma unroll
  for (int r = 0; r < 8; ++r) {
    float v = c[r] + b;
    if (relu) v = fmaxf(v, 0.f);
    out[(size_t)(i0 + r) * ldo + j] = v;
  }
}

__global__ __launch_bounds__(128) void gemm_bias_kernel(
    const unsigned short* __restrict__ A, const unsigned short* __restrict__ Bt,
    const float* __restrict__ bias, float* __restrict__ out,
    int K, int ldo, int relu) {
  const int lane = threadIdx.x & 31;
  const int wid  = threadIdx.x >> 5;          // 2x2 waves over 128x128 macro tile
  const int rBase = blockIdx.x * 128 + (wid >> 1) * 64;
  const int cBase = blockIdx.y * 128 + (wid & 1) * 64;

  const unsigned short* pa = A + (size_t)(rBase + (lane & 15)) * K + ((lane >> 4) << 3);
  const unsigned short* pb = Bt + (size_t)(cBase + (lane & 15)) * K + ((lane >> 4) << 3);
  const size_t rowStep16 = (size_t)16 * K;

  v8f acc[4][4];
#pragma unroll
  for (int i = 0; i < 4; ++i)
#pragma unroll
    for (int j = 0; j < 4; ++j) acc[i][j] = (v8f){};

  for (int k = 0; k < K; k += 32) {
    __builtin_prefetch(pa + k + 256, 0, 1);
    __builtin_prefetch(pb + k + 256, 0, 1);
    v16bf a[4], b[4];
#pragma unroll
    for (int i = 0; i < 4; ++i) a[i] = ld_a(pa + i * rowStep16 + k);
#pragma unroll
    for (int j = 0; j < 4; ++j) b[j] = ld_b(pb + j * rowStep16 + k);
#pragma unroll
    for (int i = 0; i < 4; ++i)
#pragma unroll
      for (int j = 0; j < 4; ++j)
        acc[i][j] = __builtin_amdgcn_wmma_f32_16x16x32_bf16(
            false, a[i], false, b[j], (short)0, acc[i][j], false, false);
  }

  const int iOff = (lane >> 4) << 3;
  const int jOff = lane & 15;
#pragma unroll
  for (int i = 0; i < 4; ++i)
#pragma unroll
    for (int j = 0; j < 4; ++j)
      store_tile(acc[i][j], rBase + i * 16 + iOff, cBase + j * 16 + jOff,
                 bias, out, ldo, relu);
}

// ---------------- kernel 5: discriminator tail ----------------
__global__ __launch_bounds__(64) void disc_tail_kernel(
    const float* __restrict__ h1, const float* __restrict__ dW2, const float* __restrict__ db2,
    const float* __restrict__ dW3, const float* __restrict__ db3, float* accum) {
  int row = blockIdx.x;            // 0..2559
  int j = threadIdx.x;             // 0..63
  __shared__ float h2s[64];
  const float* h = h1 + (size_t)row * 128;
  float acc = db2[j];
  for (int k = 0; k < 128; ++k) acc += h[k] * dW2[k * 64 + j];
  h2s[j] = fmaxf(acc, 0.f);
  __syncthreads();
  if (j < 2) {
    float a = db3[j];
    for (int k = 0; k < 64; ++k) a += h2s[k] * dW3[k * 2 + j];
    float s = 1.f / (1.f + expf(-a));       // sigmoid
    float sp = log1pf(expf(s));             // softplus
    atomicAdd(accum + 3, sp);
  }
}

// ---------------- kernel 6: classifier tail + preds + NLL ----------------
__global__ __launch_bounds__(32) void cls_tail_kernel(
    const float* __restrict__ c1, const float* __restrict__ cW2, const float* __restrict__ cb2,
    const float* __restrict__ cW3, const float* __restrict__ cb3,
    const int* __restrict__ label, float* __restrict__ preds, float* accum) {
  int row = blockIdx.x;            // 0..1279
  int j = threadIdx.x;             // 0..31
  __shared__ float c2s[32];
  __shared__ float ps[3];
  const float* c = c1 + (size_t)row * 128;
  float acc = cb2[j];
  for (int k = 0; k < 128; ++k) acc += c[k] * cW2[k * 32 + j];
  c2s[j] = acc;
  __syncthreads();
  if (j < 3) {
    float a = cb3[j];
    for (int k = 0; k < 32; ++k) a += c2s[k] * cW3[k * 3 + j];
    ps[j] = a;
    preds[row * 3 + j] = a;
  }
  __syncthreads();
  if (j == 0) {
    float m = fmaxf(ps[0], fmaxf(ps[1], ps[2]));
    float lse = m + logf(expf(ps[0] - m) + expf(ps[1] - m) + expf(ps[2] - m));
    int lab = label[row];
    atomicAdd(accum + 4, lse - ps[lab]);    // -log p[label]
  }
}

// ---------------- kernel 7: sum of squares (cW1 L2) ----------------
__global__ __launch_bounds__(256) void sumsq_kernel(const float* __restrict__ x, long long n, float* slot) {
  long long i = (long long)blockIdx.x * blockDim.x + threadIdx.x;
  long long stride = (long long)gridDim.x * blockDim.x;
  float s = 0.f;
  for (; i < n; i += stride) { float v = x[i]; s += v * v; }
  s = wave_sum(s);
  if ((threadIdx.x & 31) == 0) atomicAdd(slot, s);
}

// ---------------- kernel 8: finalize scalars ----------------
__global__ __launch_bounds__(256) void finalize_kernel(
    const float* __restrict__ accum,
    const float* __restrict__ cb1, const float* __restrict__ cW2, const float* __restrict__ cb2,
    const float* __restrict__ cW3, const float* __restrict__ cb3, float* __restrict__ out) {
  int tid = threadIdx.x;
  float s = 0.f;
  for (int i = tid; i < 128;  i += 256) { float v = cb1[i]; s += v * v; }
  for (int i = tid; i < 4096; i += 256) { float v = cW2[i]; s += v * v; }
  for (int i = tid; i < 32;   i += 256) { float v = cb2[i]; s += v * v; }
  for (int i = tid; i < 96;   i += 256) { float v = cW3[i]; s += v * v; }
  for (int i = tid; i < 3;    i += 256) { float v = cb3[i]; s += v * v; }
  s = wave_sum(s);
  __shared__ float red[8];
  if ((tid & 31) == 0) red[tid >> 5] = s;
  __syncthreads();
  if (tid == 0) {
    float l2 = accum[5];
    for (int w = 0; w < 8; ++w) l2 += red[w];
    float kx = accum[0], ky = accum[1], kxy = accum[2];
    const float nl = (float)NLOC;
    float lmmd = ((float)NREG / (nl * (nl - 1.f)) * kx
                  - 2.f / (nl * nl) * kxy
                  + (float)NREG / (nl * (nl - 1.f)) * ky) / ((float)NREG * (float)NREG);
    float dloss = accum[3] / (2.f * 2.f * (float)NROW);     // mean over [2560,2]
    float closs = accum[4] / (float)NROW + 0.01f * l2;
    out[3840] = closs;          // classifier_loss
    out[3841] = dloss;          // discriminator_loss
    out[3842] = dloss + closs;  // loss
    out[3843] = lmmd;           // mmd_loss
  }
}

// ---------------- host launcher ----------------
extern "C" void kernel_launch(void* const* d_in, const int* in_sizes, int n_in,
                              void* d_out, int out_size, void* d_ws, size_t ws_size,
                              hipStream_t stream) {
  const float* src = (const float*)d_in[0];
  const float* tgt = (const float*)d_in[1];
  const int*   label = (const int*)d_in[2];
  const float* dW1 = (const float*)d_in[3];
  const float* db1 = (const float*)d_in[4];
  const float* dW2 = (const float*)d_in[5];
  const float* db2 = (const float*)d_in[6];
  const float* dW3 = (const float*)d_in[7];
  const float* db3 = (const float*)d_in[8];
  const float* cW1 = (const float*)d_in[9];
  const float* cb1 = (const float*)d_in[10];
  const float* cW2 = (const float*)d_in[11];
  const float* cb2 = (const float*)d_in[12];
  const float* cW3 = (const float*)d_in[13];
  const float* cb3 = (const float*)d_in[14];
  float* out = (float*)d_out;
  char* ws = (char*)d_ws;

  // workspace layout (bytes)
  unsigned short* sB   = (unsigned short*)(ws);                     // 1280*32768*2 = 83,886,080
  unsigned short* tB   = (unsigned short*)(ws + 83886080ull);       // + same
  unsigned short* dW1t = (unsigned short*)(ws + 167772160ull);      // 128*32768*2 = 8,388,608
  unsigned short* cW1t = (unsigned short*)(ws + 176160768ull);      // + same
  float* s2    = (float*)(ws + 184549376ull);                       // 1280*4
  float* t2    = (float*)(ws + 184554496ull);                       // 1280*4
  float* h1    = (float*)(ws + 184559616ull);                       // 2560*128*4
  float* c1    = (float*)(ws + 185870336ull);                       // 1280*128*4
  float* accum = (float*)(ws + 186525696ull);                       // 8 floats

  zero_accum_kernel<<<1, 32, 0, stream>>>(accum);

  feat_prep_kernel<<<2 * NROW, 256, 0, stream>>>(src, tgt, sB, tB, s2, t2);

  transpose_convert_kernel<<<dim3(FDIM / 32, 4), dim3(32, 8), 0, stream>>>(dW1, dW1t, FDIM, 128);
  transpose_convert_kernel<<<dim3(FDIM / 32, 4), dim3(32, 8), 0, stream>>>(cW1, cW1t, FDIM, 128);

  // Kxy: full 1280x1280 cross kernel (128x128 macro tiles)
  gram_exp_kernel<<<dim3(NROW / 128, NROW / 128, 1), 128, 0, stream>>>(
      sB, tB, s2, t2, FDIM, 0, 0, accum + 2);
  // Kx: block-diagonal per-region grams (one 128x128 block per region)
  gram_exp_kernel<<<dim3(1, 1, NREG), 128, 0, stream>>>(
      sB, sB, s2, s2, FDIM, NLOC, 1, accum + 0);
  // Ky
  gram_exp_kernel<<<dim3(1, 1, NREG), 128, 0, stream>>>(
      tB, tB, t2, t2, FDIM, NLOC, 1, accum + 1);

  // disc layer1 over dom = [src;tgt] (sB,tB contiguous -> 2560 rows), relu
  gemm_bias_kernel<<<dim3(2 * NROW / 128, 1), 128, 0, stream>>>(
      sB, dW1t, db1, h1, FDIM, 128, 1);
  // cls layer1 over tgt, no activation
  gemm_bias_kernel<<<dim3(NROW / 128, 1), 128, 0, stream>>>(
      tB, cW1t, cb1, c1, FDIM, 128, 0);

  disc_tail_kernel<<<2 * NROW, 64, 0, stream>>>(h1, dW2, db2, dW3, db3, accum);
  cls_tail_kernel<<<NROW, 32, 0, stream>>>(c1, cW2, cb2, cW3, cb3, label, out, accum);

  sumsq_kernel<<<256, 256, 0, stream>>>(cW1, (long long)FDIM * 128, accum + 5);

  finalize_kernel<<<1, 256, 0, stream>>>(accum, cb1, cW2, cb2, cW3, cb3, out);
}
